// SPANet_15187004358991
// MI455X (gfx1250) — matrix operand order repaired and checked by reading, compiled, not run on hardware
//
#include <hip/hip_runtime.h>

// ---------------------------------------------------------------------------
// SPANet on MI455X (gfx1250): NHWC f16 activations, implicit-GEMM convs via
// v_wmma_f32_16x16x32_f16, f32 accumulation, fused epilogues.
// Weights are pre-permuted into per-lane WMMA A-fragment order so each A
// fragment is one aligned 32-byte LDS read (no packing VALU).
// ---------------------------------------------------------------------------

typedef _Float16 v8h  __attribute__((ext_vector_type(8)));
typedef _Float16 v16h __attribute__((ext_vector_type(16)));
typedef float    v8f  __attribute__((ext_vector_type(8)));

#define H 256
#define W 256
#define NB 4
#define NPIX (NB * H * W)   // 262144

// ---------------------------------------------------------------------------
// Weight conversion: OIHW f32 -> f16, permuted into A-fragment order:
//   dst[(((g*taps + tap)*2 + m)*32 + lane)*16 + k]
//     = W[co = m*16 + (lane&15)][ci = g*32 + K0 + k][ky][kx]
//   with K0 = (lane<16) ? 0 : 16  (contiguous k order, matches B loads)
// so conv kernel's A load is a single contiguous 32B LDS read per lane.
// ---------------------------------------------------------------------------
__global__ void wcvt_k(const float* __restrict__ w, _Float16* __restrict__ dst,
                       int cin, int ks) {
  int taps  = ks * ks;
  int total = 32 * cin * taps;
  int i = blockIdx.x * 256 + threadIdx.x;
  if (i >= total) return;
  int k    = i & 15;
  int lane = (i >> 4) & 31;
  int m    = (i >> 9) & 1;
  int rest = i >> 10;                 // g*taps + tap
  int tap  = rest % taps;
  int g    = rest / taps;
  int K0   = (lane < 16) ? 0 : 16;
  int cil  = K0 + k;   // K0..K0+15
  int co   = m * 16 + (lane & 15);
  int ci   = g * 32 + cil;
  int ky   = tap / ks, kx = tap % ks;
  dst[i] = (_Float16)w[((co * cin + ci) * ks + ky) * ks + kx];
}

// ---------------------------------------------------------------------------
// Implicit-GEMM conv: wave32 computes a 32-pixel strip x 32 Cout tile
// (two 16-pixel B strips reuse each A fragment -> 4 WMMAs per A pair).
//   A (16x32 f16)  = weights from LDS, one v16h ds read per fragment
//   B (32x16 f16)  = activations, NHWC global, one 32B contiguous lane load
//   C (16x16 f32)  = v8f accumulators
// fuse: 0 = none, 1 = relu, 2 = relu(acc * att + skip)   (att==null -> att=1)
// ---------------------------------------------------------------------------
template <int CIN, int KS>
__global__ __launch_bounds__(128)
void conv_wmma_k(const _Float16* __restrict__ in, const _Float16* __restrict__ w16,
                 _Float16* __restrict__ out, int pad, int dil, int fuse,
                 const float* __restrict__ att, const _Float16* __restrict__ skip) {
  constexpr int CIG  = CIN / 32;
  constexpr int TAPS = KS * KS;
  __shared__ __align__(32) _Float16 wlds[TAPS * 1024];  // one ci-group, all taps

  const int lane = threadIdx.x & 31;
  const int wave = threadIdx.x >> 5;
  const int tile = blockIdx.x * 4 + wave;        // 8192 tiles of 32 pixels
  const int x0   = (tile & 7) << 5;
  const int y    = (tile >> 3) & 255;
  const int n    = tile >> 11;
  const int pxA  = x0 + (lane & 15);
  const int pxB  = pxA + 16;
  const int hi   = (lane < 16) ? 0 : 1;          // half select

  v8f c0A = {}, c1A = {}, c0B = {}, c1B = {};

  for (int g = 0; g < CIG; ++g) {
    __syncthreads();
    { // cooperative weight stage for this ci-group: TAPS*1024 halves
      const v8h* src = (const v8h*)(w16 + g * TAPS * 1024);
      v8h* dst = (v8h*)wlds;
      for (int i = threadIdx.x; i < TAPS * 128; i += 128) dst[i] = src[i];
    }
    __syncthreads();

#pragma unroll
    for (int tap = 0; tap < TAPS; ++tap) {
      const int dy = tap / KS, dx = tap % KS;
      const int yi  = y + dy * dil - pad;
      const int xiA = pxA + dx * dil - pad;
      const int xiB = pxB + dx * dil - pad;
      const bool okY = (unsigned)yi < (unsigned)H;

      v16h bA = {}, bB = {};
      if (okY & ((unsigned)xiA < (unsigned)W))
        bA = *(const v16h*)(in + ((n * H + yi) * W + xiA) * CIN + g * 32 + hi * 16);
      if (okY & ((unsigned)xiB < (unsigned)W))
        bB = *(const v16h*)(in + ((n * H + yi) * W + xiB) * CIN + g * 32 + hi * 16);

      // A fragments: one aligned 32B LDS read each (pre-permuted layout)
      const v16h a0 = *(const v16h*)(wlds + tap * 1024 + lane * 16);
      const v16h a1 = *(const v16h*)(wlds + tap * 1024 + 512 + lane * 16);

      c0A = __builtin_amdgcn_wmma_f32_16x16x32_f16(false, a0, false, bA, (short)0, c0A, false, false);
      c1A = __builtin_amdgcn_wmma_f32_16x16x32_f16(false, a1, false, bA, (short)0, c1A, false, false);
      c0B = __builtin_amdgcn_wmma_f32_16x16x32_f16(false, a0, false, bB, (short)0, c0B, false, false);
      c1B = __builtin_amdgcn_wmma_f32_16x16x32_f16(false, a1, false, bB, (short)0, c1B, false, false);
    }
  }

  // Epilogue. C/D layout: lane<16 -> co r (c0), co 16+r (c1); lane>=16 -> 8+r / 24+r.
  auto epi = [&](int px, const v8f& c0, const v8f& c1) {
    const int pix = (n * H + y) * W + px;
    _Float16* op = out + pix * 32 + hi * 8;
    v8h o0, o1;
    if (fuse == 2) {
      const float scale = att ? att[pix] : 1.0f;
      const v8h s0 = *(const v8h*)(skip + pix * 32 + hi * 8);
      const v8h s1 = *(const v8h*)(skip + pix * 32 + 16 + hi * 8);
#pragma unroll
      for (int i = 0; i < 8; ++i) {
        o0[i] = (_Float16)fmaxf(fmaf(c0[i], scale, (float)s0[i]), 0.0f);
        o1[i] = (_Float16)fmaxf(fmaf(c1[i], scale, (float)s1[i]), 0.0f);
      }
    } else if (fuse == 1) {
#pragma unroll
      for (int i = 0; i < 8; ++i) {
        o0[i] = (_Float16)fmaxf(c0[i], 0.0f);
        o1[i] = (_Float16)fmaxf(c1[i], 0.0f);
      }
    } else {
#pragma unroll
      for (int i = 0; i < 8; ++i) { o0[i] = (_Float16)c0[i]; o1[i] = (_Float16)c1[i]; }
    }
    *(v8h*)op = o0;
    *(v8h*)(op + 16) = o1;
  };
  epi(pxA, c0A, c1A);
  epi(pxB, c0B, c1B);
}

// ---------------------------------------------------------------------------
// Stem: 3->32 3x3 pad1 + relu. NCHW f32 in, NHWC f16 out. Thread per pixel.
// ---------------------------------------------------------------------------
__global__ __launch_bounds__(256)
void conv_in_k(const float* __restrict__ x, const float* __restrict__ w,
               _Float16* __restrict__ out) {
  __shared__ float wl[32 * 3 * 9];   // OIHW flat: [co][ci][tap]
  for (int i = threadIdx.x; i < 864; i += 256) wl[i] = w[i];
  __syncthreads();
  int pid = blockIdx.x * 256 + threadIdx.x;
  int px = pid & 255, y = (pid >> 8) & 255, n = pid >> 16;
  float acc[32];
#pragma unroll
  for (int c = 0; c < 32; ++c) acc[c] = 0.0f;
  for (int ci = 0; ci < 3; ++ci) {
#pragma unroll
    for (int tap = 0; tap < 9; ++tap) {
      int yi = y + tap / 3 - 1, xi = px + tap % 3 - 1;
      if ((unsigned)yi < (unsigned)H && (unsigned)xi < (unsigned)W) {
        float v = x[((n * 3 + ci) * H + yi) * W + xi];
#pragma unroll
        for (int co = 0; co < 32; ++co) acc[co] = fmaf(v, wl[(co * 3 + ci) * 9 + tap], acc[co]);
      }
    }
  }
  _Float16* o = out + pid * 32;
#pragma unroll
  for (int c = 0; c < 4; ++c) {
    v8h v;
#pragma unroll
    for (int j = 0; j < 8; ++j) v[j] = (_Float16)fmaxf(acc[c * 8 + j], 0.0f);
    *(v8h*)(o + c * 8) = v;
  }
}

// ---------------------------------------------------------------------------
// Head: 32->3 3x3 pad1. NHWC f16 in, NCHW f32 out (into d_out). Thread/pixel.
// ---------------------------------------------------------------------------
__global__ __launch_bounds__(256)
void conv_out_k(const _Float16* __restrict__ in, const float* __restrict__ w,
                float* __restrict__ outp) {
  __shared__ float wl[3 * 32 * 9];   // [co][ci][tap]
  for (int i = threadIdx.x; i < 864; i += 256) wl[i] = w[i];
  __syncthreads();
  int pid = blockIdx.x * 256 + threadIdx.x;
  int px = pid & 255, y = (pid >> 8) & 255, n = pid >> 16;
  float acc0 = 0, acc1 = 0, acc2 = 0;
#pragma unroll
  for (int tap = 0; tap < 9; ++tap) {
    int yi = y + tap / 3 - 1, xi = px + tap % 3 - 1;
    if ((unsigned)yi < (unsigned)H && (unsigned)xi < (unsigned)W) {
      const v8h* p = (const v8h*)(in + ((n * H + yi) * W + xi) * 32);
#pragma unroll
      for (int c = 0; c < 4; ++c) {
        v8h v = p[c];
#pragma unroll
        for (int j = 0; j < 8; ++j) {
          int ci = c * 8 + j;
          float xv = (float)v[j];
          acc0 = fmaf(xv, wl[(0 * 32 + ci) * 9 + tap], acc0);
          acc1 = fmaf(xv, wl[(1 * 32 + ci) * 9 + tap], acc1);
          acc2 = fmaf(xv, wl[(2 * 32 + ci) * 9 + tap], acc2);
        }
      }
    }
  }
  outp[((n * 3 + 0) * H + y) * W + px] = acc0;
  outp[((n * 3 + 1) * H + y) * W + px] = acc1;
  outp[((n * 3 + 2) * H + y) * W + px] = acc2;
}

// ---------------------------------------------------------------------------
// IRNN scan: h_t = relu(w_c*h_{t-1} + b_c + x_t), out[0]=x[0]. NHWC coalesced:
// one wave reads/writes one pixel's 32 channels per step. Writes into the
// 128-channel concat buffer at catOff.
// ---------------------------------------------------------------------------
__global__ __launch_bounds__(256)
void irnn_scan_k(const _Float16* __restrict__ in, _Float16* __restrict__ cat,
                 const float* __restrict__ w, const float* __restrict__ b,
                 int axis, int rev, int catOff) {
  int tid = blockIdx.x * 256 + threadIdx.x;  // 32768 chains
  int c = tid & 31;
  int q = (tid >> 5) & 255;                  // fixed coordinate (x for axis0, y for axis1)
  int n = tid >> 13;
  float wc = w[c], bc = b[c];
  int base, step;
  if (axis == 0) { base = (n * H + 0) * W + q; step = W; }   // scan over y
  else           { base = (n * H + q) * W + 0; step = 1; }   // scan over x
  int t0 = rev ? 255 : 0;
  int dt = rev ? -step : step;
  int pix = base + t0 * step;
  float h = (float)in[pix * 32 + c];
  cat[pix * 128 + catOff + c] = (_Float16)h;
  for (int s = 1; s < 256; ++s) {
    pix += dt;
    float xv = (float)in[pix * 32 + c];
    h = fmaxf(fmaf(wc, h, bc + xv), 0.0f);
    cat[pix * 128 + catOff + c] = (_Float16)h;
  }
}

// ---------------------------------------------------------------------------
// SAM gate: 1x1 conv 32->1 + sigmoid. Optionally also writes A1 to d_out.
// ---------------------------------------------------------------------------
__global__ __launch_bounds__(256)
void sam_out_k(const _Float16* __restrict__ in, const float* __restrict__ w32,
               float* __restrict__ A, float* __restrict__ A2) {
  __shared__ float wl[32];
  if (threadIdx.x < 32) wl[threadIdx.x] = w32[threadIdx.x];
  __syncthreads();
  int pid = blockIdx.x * 256 + threadIdx.x;
  const v8h* p = (const v8h*)(in + pid * 32);
  float acc = 0.0f;
#pragma unroll
  for (int c = 0; c < 4; ++c) {
    v8h v = p[c];
#pragma unroll
    for (int j = 0; j < 8; ++j) acc = fmaf((float)v[j], wl[c * 8 + j], acc);
  }
  float s = 1.0f / (1.0f + __expf(-acc));
  A[pid] = s;
  if (A2) A2[pid] = s;
}

// ---------------------------------------------------------------------------
// Host orchestration
// ---------------------------------------------------------------------------
extern "C" void kernel_launch(void* const* d_in, const int* in_sizes, int n_in,
                              void* d_out, int out_size, void* d_ws, size_t ws_size,
                              hipStream_t stream) {
  (void)in_sizes; (void)n_in; (void)out_size; (void)ws_size;

  // JAX pytree flattening (dict keys sorted):
  const float* blkW[17][3];
  for (int b = 0; b < 17; ++b)
    for (int j = 0; j < 3; ++j) blkW[b][j] = (const float*)d_in[3 * b + j];
  const float* w_conv_in  = (const float*)d_in[51];
  const float* w_conv_out = (const float*)d_in[52];
  const float* w_sam_c2   = (const float*)d_in[53];  // (32,128,3,3)
  const float* w_sam_c3   = (const float*)d_in[54];  // (32,128,3,3)
  const float* w_sam_ci   = (const float*)d_in[55];  // (32,32,3,3)
  const float* w_sam_co   = (const float*)d_in[56];  // (1,32,1,1)
  const float* irnn[2][8]; // sorted: down_b, down_w, left_b, left_w, right_b, right_w, up_b, up_w
  for (int r = 0; r < 2; ++r)
    for (int k = 0; k < 8; ++k) irnn[r][k] = (const float*)d_in[57 + 8 * r + k];
  const float* x = (const float*)d_in[73];

  // Workspace layout
  char* ws = (char*)d_ws;
  auto wsAlloc = [&](size_t bytes) -> void* {
    void* p = ws; ws += (bytes + 255) & ~(size_t)255; return p;
  };
  _Float16* actA = (_Float16*)wsAlloc((size_t)NPIX * 32 * 2);
  _Float16* actB = (_Float16*)wsAlloc((size_t)NPIX * 32 * 2);
  _Float16* h1   = (_Float16*)wsAlloc((size_t)NPIX * 32 * 2);
  _Float16* h2   = (_Float16*)wsAlloc((size_t)NPIX * 32 * 2);
  _Float16* samO = (_Float16*)wsAlloc((size_t)NPIX * 32 * 2);
  _Float16* cat  = (_Float16*)wsAlloc((size_t)NPIX * 128 * 2);
  float*    Abuf = (float*)wsAlloc((size_t)NPIX * 4);
  _Float16 *wc1[17], *wc2[17], *wc3[17];
  for (int b = 0; b < 17; ++b) {
    wc1[b] = (_Float16*)wsAlloc(32 * 32 * 2);
    wc2[b] = (_Float16*)wsAlloc(32 * 32 * 9 * 2);
    wc3[b] = (_Float16*)wsAlloc(32 * 32 * 2);
  }
  _Float16* wsamci = (_Float16*)wsAlloc(32 * 32 * 9 * 2);
  _Float16* wsam2  = (_Float16*)wsAlloc(32 * 128 * 9 * 2);
  _Float16* wsam3  = (_Float16*)wsAlloc(32 * 128 * 9 * 2);

  auto cvt = [&](const float* src, _Float16* dst, int cin, int ks) {
    int total = 32 * cin * ks * ks;
    wcvt_k<<<(total + 255) / 256, 256, 0, stream>>>(src, dst, cin, ks);
  };
  for (int b = 0; b < 17; ++b) {
    cvt(blkW[b][0], wc1[b], 32, 1);
    cvt(blkW[b][1], wc2[b], 32, 3);
    cvt(blkW[b][2], wc3[b], 32, 1);
  }
  cvt(w_sam_ci, wsamci, 32, 3);
  cvt(w_sam_c2, wsam2, 128, 3);
  cvt(w_sam_c3, wsam3, 128, 3);

  const dim3 CG(2048), CB(128);   // 8192 wave-tiles of 32 pixels

  conv_in_k<<<NPIX / 256, 256, 0, stream>>>(x, w_conv_in, actA);

  _Float16* cur = actA;
  _Float16* nxt = actB;
  auto bneck = [&](int b, const float* att) {
    conv_wmma_k<32, 1><<<CG, CB, 0, stream>>>(cur, wc1[b], h1, 0, 1, 1, nullptr, nullptr);
    conv_wmma_k<32, 3><<<CG, CB, 0, stream>>>(h1, wc2[b], h2, 2, 2, 1, nullptr, nullptr);
    conv_wmma_k<32, 1><<<CG, CB, 0, stream>>>(h2, wc3[b], nxt, 0, 1, 2, att, cur);
    _Float16* t = cur; cur = nxt; nxt = t;
  };
  auto irnnAll = [&](int r) {  // concat order: up@0, right@32, down@64, left@96
    irnn_scan_k<<<128, 256, 0, stream>>>(samO, cat, irnn[r][7], irnn[r][6], 0, 1, 0);
    irnn_scan_k<<<128, 256, 0, stream>>>(samO, cat, irnn[r][5], irnn[r][4], 1, 0, 32);
    irnn_scan_k<<<128, 256, 0, stream>>>(samO, cat, irnn[r][1], irnn[r][0], 0, 0, 64);
    irnn_scan_k<<<128, 256, 0, stream>>>(samO, cat, irnn[r][3], irnn[r][2], 1, 1, 96);
  };
  auto sam = [&](float* Adst2) {
    conv_wmma_k<32, 3><<<CG, CB, 0, stream>>>(cur, wsamci, samO, 1, 1, 0, nullptr, nullptr);
    irnnAll(0);
    conv_wmma_k<128, 3><<<CG, CB, 0, stream>>>(cat, wsam2, samO, 1, 1, 0, nullptr, nullptr);
    irnnAll(1);
    conv_wmma_k<128, 3><<<CG, CB, 0, stream>>>(cat, wsam3, samO, 1, 1, 1, nullptr, nullptr);
    sam_out_k<<<NPIX / 256, 256, 0, stream>>>(samO, w_sam_co, Abuf, Adst2);
  };

  for (int b = 0; b < 3; ++b) bneck(b, nullptr);
  sam((float*)d_out);                       // A1 -> first 262144 floats of d_out
  for (int b = 3; b < 6; ++b) bneck(b, Abuf);
  sam(nullptr);
  for (int b = 6; b < 9; ++b) bneck(b, Abuf);
  sam(nullptr);
  for (int b = 9; b < 12; ++b) bneck(b, Abuf);
  sam(nullptr);
  for (int b = 12; b < 15; ++b) bneck(b, Abuf);
  bneck(15, nullptr);
  bneck(16, nullptr);

  conv_out_k<<<NPIX / 256, 256, 0, stream>>>(cur, w_conv_out, (float*)d_out + NPIX);
}
